// dygcc_dype_v3_Block_81346680586345
// MI455X (gfx1250) — matrix operand order repaired and checked by reading, not compiled, not run on hardware
//
#include <hip/hip_runtime.h>
#include <hip/hip_bf16.h>

// ---------------------------------------------------------------------------
// Shapes (fixed by the reference): B=16, DIM=512, H=W=32, MID=32
// ---------------------------------------------------------------------------
#define BATCH 16
#define DIM   512
#define MID   32
#define HW    32
#define NPIX  (BATCH * HW * HW)     // 16384
#define FF    (4 * DIM)             // 2048

typedef __attribute__((ext_vector_type(16))) _Float16 v16h;
typedef __attribute__((ext_vector_type(8)))  _Float16 v8h;
typedef __attribute__((ext_vector_type(8)))  float    v8f;

union HFrag { v16h v; v8h h[2]; };

// ---------------------------------------------------------------------------
// Conv 3x3, 512 -> 32, fused BN + ReLU.  One block per (b, h) row.
// ---------------------------------------------------------------------------
__global__ __launch_bounds__(256) void conv_reduce_bn_relu(
    const float* __restrict__ in,    // B x 512 x 32 x 32
    const float* __restrict__ wgt,   // 32 x 512 x 3 x 3
    const float* __restrict__ bng, const float* __restrict__ bnb,
    const float* __restrict__ bnm, const float* __restrict__ bnv,
    float* __restrict__ outp)        // B x 32 x 32 x 32
{
    const int b = blockIdx.x >> 5;
    const int h = blockIdx.x & 31;
    __shared__ float xs[64 * 3 * 36];      // 64 in-ch chunk, 3 rows, 36 cols
    const int t = threadIdx.x;
    const int wcol = t & 31;
    const int oc0  = (t >> 5) * 4;         // 8 groups x 4 oc = 32 oc
    float acc[4] = {0.f, 0.f, 0.f, 0.f};

    for (int c0 = 0; c0 < DIM; c0 += 64) {
        __syncthreads();
        #pragma unroll 1
        for (int k = 0; k < 27; ++k) {     // 6912 = 256*27 elements
            int idx = t + k * 256;
            int ch  = idx / 108;
            int rem = idx - ch * 108;
            int r   = rem / 36;
            int xc  = rem - r * 36;
            int iw  = xc - 1;
            int row = h - 1 + r;
            float v = 0.f;
            if ((unsigned)iw < 32u && (unsigned)row < 32u)
                v = in[(((size_t)b * DIM + c0 + ch) * HW + row) * HW + iw];
            xs[(ch * 3 + r) * 36 + xc] = v;
        }
        __syncthreads();
        #pragma unroll 1
        for (int ch = 0; ch < 64; ++ch) {
            const float* wp = wgt + ((size_t)oc0 * DIM + (c0 + ch)) * 9;
            #pragma unroll
            for (int r = 0; r < 3; ++r) {
                #pragma unroll
                for (int dx = 0; dx < 3; ++dx) {
                    float xv = xs[(ch * 3 + r) * 36 + wcol + dx];
                    #pragma unroll
                    for (int k2 = 0; k2 < 4; ++k2)
                        acc[k2] = fmaf(xv, wp[(size_t)k2 * DIM * 9 + r * 3 + dx], acc[k2]);
                }
            }
        }
    }
    #pragma unroll
    for (int k2 = 0; k2 < 4; ++k2) {
        int oc = oc0 + k2;
        float inv = rsqrtf(bnv[oc] + 1e-5f);
        float v = bng[oc] * (acc[k2] - bnm[oc]) * inv + bnb[oc];
        v = v > 0.f ? v : 0.f;
        outp[(((size_t)b * MID + oc) * HW + h) * HW + wcol] = v;
    }
}

// ---------------------------------------------------------------------------
// Conv 3x3, 32 -> 512, fused +bias (+ optional residual).
// One block per (b, h, group-of-64-out-channels).
// ---------------------------------------------------------------------------
__global__ __launch_bounds__(256) void conv_expand(
    const float* __restrict__ in,    // B x 32 x 32 x 32  (mid activations)
    const float* __restrict__ wgt,   // 512 x 32 x 3 x 3
    const float* __restrict__ b2,    // 512
    const float* __restrict__ resid, // B x 512 x 32 x 32 or nullptr
    float* __restrict__ outp)        // B x 512 x 32 x 32
{
    const int blk = blockIdx.x;
    const int g = blk & 7;
    const int h = (blk >> 3) & 31;
    const int b = blk >> 8;
    __shared__ float xs[32 * 3 * 36];      // 3456 floats
    const int t = threadIdx.x;
    for (int idx = t; idx < 3456; idx += 256) {
        int ch  = idx / 108;
        int rem = idx - ch * 108;
        int r   = rem / 36;
        int xc  = rem - r * 36;
        int iw  = xc - 1;
        int row = h - 1 + r;
        float v = 0.f;
        if ((unsigned)iw < 32u && (unsigned)row < 32u)
            v = in[(((size_t)b * MID + ch) * HW + row) * HW + iw];
        xs[(ch * 3 + r) * 36 + xc] = v;
    }
    __syncthreads();
    const int wcol = t & 31;
    const int oc0  = g * 64 + (t >> 5) * 8;
    float acc[8] = {};
    #pragma unroll 1
    for (int ch = 0; ch < MID; ++ch) {
        #pragma unroll
        for (int r = 0; r < 3; ++r) {
            #pragma unroll
            for (int dx = 0; dx < 3; ++dx) {
                float xv = xs[(ch * 3 + r) * 36 + wcol + dx];
                const float* wp = wgt + (((size_t)oc0 * MID + ch) * 3 + r) * 3 + dx;
                #pragma unroll
                for (int k = 0; k < 8; ++k)
                    acc[k] = fmaf(xv, wp[(size_t)k * MID * 9], acc[k]);
            }
        }
    }
    #pragma unroll
    for (int k = 0; k < 8; ++k) {
        int oc = oc0 + k;
        size_t oi = (((size_t)b * DIM + oc) * HW + h) * HW + wcol;
        float v = acc[k] + b2[oc];
        if (resid) v += resid[oi];
        outp[oi] = v;
    }
}

// ---------------------------------------------------------------------------
// Hw = mean_W kw[:, :256];  Ww = mean_H kw[:, 256:]
// ---------------------------------------------------------------------------
__global__ __launch_bounds__(256) void rowcol_mean(
    const float* __restrict__ kw, float* __restrict__ Hw, float* __restrict__ Ww)
{
    int gid = blockIdx.x * 256 + threadIdx.x;          // 2 * 16*256*32 = 262144
    if (gid < 131072) {
        int b = gid >> 13; int c = (gid >> 5) & 255; int i = gid & 31;
        const float* p = kw + (((size_t)b * DIM + c) * HW + i) * HW;
        float s = 0.f;
        #pragma unroll
        for (int w = 0; w < 32; ++w) s += p[w];
        Hw[gid] = s * (1.f / 32.f);
    } else {
        int id = gid - 131072;
        int b = id >> 13; int c = (id >> 5) & 255; int i = id & 31;
        const float* p = kw + (((size_t)b * DIM + 256 + c) * HW) * HW + i;
        float s = 0.f;
        #pragma unroll
        for (int h = 0; h < 32; ++h) s += p[h * 32];
        Ww[id] = s * (1.f / 32.f);
    }
}

// ---------------------------------------------------------------------------
// Circulant mixing + bias, writes NHWC fp32.  gid == NHWC output index.
// ---------------------------------------------------------------------------
__global__ __launch_bounds__(256) void circulant_mix(
    const float* __restrict__ x1, const float* __restrict__ Hw,
    const float* __restrict__ Ww, const float* __restrict__ bias,
    float* __restrict__ ynhwc)
{
    int gid = blockIdx.x * 256 + threadIdx.x;          // 16*32*32*512 = 8388608
    int c = gid & 511;
    int q = (gid >> 9) & 31;
    int p = (gid >> 14) & 31;
    int b = gid >> 19;
    float acc = 0.f;
    if (c < 256) {
        // y1[b,c,i=p,w=q] = sum_j Hw[b,c,(j-p)%32] * x1[b,c,j,q]
        const float* Hr = Hw + ((size_t)b * 256 + c) * 32;
        const float* xp = x1 + (((size_t)b * DIM + c) * HW) * HW + q;
        #pragma unroll
        for (int j = 0; j < 32; ++j)
            acc = fmaf(Hr[(j - p) & 31], xp[j * 32], acc);
    } else {
        // y2[b,c,h=p,i=q] = sum_j Ww[b,c-256,(j-q)%32] * x1[b,c,p,j]
        const float* Wr = Ww + ((size_t)b * 256 + (c - 256)) * 32;
        const float* xp = x1 + (((size_t)b * DIM + c) * HW + p) * HW;
        #pragma unroll
        for (int j = 0; j < 32; ++j)
            acc = fmaf(Wr[(j - q) & 31], xp[j], acc);
    }
    ynhwc[gid] = acc + bias[c];
}

// ---------------------------------------------------------------------------
// LayerNorm over 512 channels per pixel (one wave32 per pixel), emits f16.
// ---------------------------------------------------------------------------
__global__ __launch_bounds__(256) void layernorm_f16(
    const float* __restrict__ y, const float* __restrict__ lw,
    const float* __restrict__ lb, _Float16* __restrict__ outh)
{
    const int wv   = threadIdx.x >> 5;
    const int lane = threadIdx.x & 31;
    const int pix  = blockIdx.x * 8 + wv;
    const float* p = y + (size_t)pix * DIM;
    float vbuf[16];
    float s = 0.f, ss = 0.f;
    #pragma unroll
    for (int i = 0; i < 16; ++i) {
        float v = p[lane + i * 32];
        vbuf[i] = v; s += v; ss += v * v;
    }
    #pragma unroll
    for (int o = 16; o >= 1; o >>= 1) {
        s  += __shfl_xor(s,  o, 32);
        ss += __shfl_xor(ss, o, 32);
    }
    float mu  = s * (1.f / 512.f);
    float var = ss * (1.f / 512.f) - mu * mu;
    float rs  = rsqrtf(var + 1e-6f);
    _Float16* q = outh + (size_t)pix * DIM;
    #pragma unroll
    for (int i = 0; i < 16; ++i) {
        int cch = lane + i * 32;
        q[cch] = (_Float16)((vbuf[i] - mu) * rs * lw[cch] + lb[cch]);
    }
}

// ---------------------------------------------------------------------------
// fp32 -> f16 weight conversion
// ---------------------------------------------------------------------------
__global__ __launch_bounds__(256) void cvt_f32_to_f16(
    const float* __restrict__ src, _Float16* __restrict__ dst, int n)
{
    int i = blockIdx.x * 256 + threadIdx.x;
    if (i < n) dst[i] = (_Float16)src[i];
}

// ---------------------------------------------------------------------------
// WMMA GEMM: C[M x N] = A[M x K] * Bt[N x K]^T  (f16 in, f32 accumulate)
// Block: 256 threads (8 waves), tile 64(M) x 128(N), K-step 32.
// Each wave: 32x32 patch -> 4x v_wmma_f32_16x16x32_f16 per K-step.
// MODE 0: +bias, exact GELU, f16 out (stride FF).
// MODE 1: +bias, out = x_orig + gamma * v, fp32 NCHW out.
// ---------------------------------------------------------------------------
template <int KDIM, int MODE>
__global__ __launch_bounds__(256) void gemm_wmma(
    const _Float16* __restrict__ A,   // M x KDIM, row-major
    const _Float16* __restrict__ Bt,  // N x KDIM, row-major (n-major B)
    const float* __restrict__ bias,   // N
    const float* __restrict__ x_orig, // MODE 1: original input, NCHW
    const float* __restrict__ gamma,  // MODE 1
    _Float16* __restrict__ Ch,        // MODE 0 output
    float* __restrict__ Cf)           // MODE 1 output
{
    __shared__ _Float16 As[64 * 40];    // 64 rows, pitch 40 halfs (80B, 16B-aligned)
    __shared__ _Float16 Bs[128 * 40];   // 128 n-rows, pitch 40 halfs

    const int tid  = threadIdx.x;
    const int lane = tid & 31;
    const int wv   = tid >> 5;
    const int wm   = (wv >> 2) * 32;    // wave M offset within block tile
    const int wn   = (wv & 3) * 32;     // wave N offset within block tile
    const int m0   = blockIdx.y * 64;
    const int n0   = blockIdx.x * 128;

    v8f acc[2][2];
    #pragma unroll
    for (int i = 0; i < 2; ++i)
        #pragma unroll
        for (int j = 0; j < 2; ++j)
            #pragma unroll
            for (int r = 0; r < 8; ++r) acc[i][j][r] = 0.f;

    const int arow = tid >> 2, ac4 = tid & 3;          // A cooperative load map
    for (int kk = 0; kk < KDIM; kk += 32) {
        // ---- stage A tile (64 x 32) ----
        *reinterpret_cast<v8h*>(&As[arow * 40 + ac4 * 8]) =
            *reinterpret_cast<const v8h*>(A + (size_t)(m0 + arow) * KDIM + kk + ac4 * 8);
        // ---- stage B tile (128 n-rows x 32 k) ----
        #pragma unroll
        for (int j = 0; j < 2; ++j) {
            int id = tid + j * 256;
            int rn = id >> 2, d4 = id & 3;
            *reinterpret_cast<v8h*>(&Bs[rn * 40 + d4 * 8]) =
                *reinterpret_cast<const v8h*>(Bt + (size_t)(n0 + rn) * KDIM + kk + d4 * 8);
        }
        if (kk + 32 < KDIM) {   // hint next A tile -> global_prefetch_b8
            __builtin_prefetch(A + (size_t)(m0 + arow) * KDIM + kk + 32 + ac4 * 8, 0, 1);
        }
        __syncthreads();

        // ---- build fragments per ISA 16-bit layouts ----
        const int ao = (lane & 16) ? 8 : 0;    // A: lanes16-31 hold K+8 / K+24
        const int kb = (lane & 16) ? 16 : 0;   // unused; B uses ao too: lanes16-31 hold K+8 / K+24
        HFrag a0, a1, b0, b1;
        const _Float16* ar = &As[(wm + (lane & 15)) * 40];
        a0.h[0] = *reinterpret_cast<const v8h*>(ar + ao);
        a0.h[1] = *reinterpret_cast<const v8h*>(ar + 16 + ao);
        a1.h[0] = *reinterpret_cast<const v8h*>(ar + 16 * 40 + ao);
        a1.h[1] = *reinterpret_cast<const v8h*>(ar + 16 * 40 + 16 + ao);
        const _Float16* br = &Bs[(wn + (lane & 15)) * 40];
        b0.h[0] = *reinterpret_cast<const v8h*>(br + ao);
        b0.h[1] = *reinterpret_cast<const v8h*>(br + 16 + ao);
        b1.h[0] = *reinterpret_cast<const v8h*>(br + 16 * 40 + ao);
        b1.h[1] = *reinterpret_cast<const v8h*>(br + 16 * 40 + 16 + ao);

        acc[0][0] = __builtin_amdgcn_wmma_f32_16x16x32_f16(
            false, a0.v, false, b0.v, (short)0, acc[0][0], false, false);
        acc[0][1] = __builtin_amdgcn_wmma_f32_16x16x32_f16(
            false, a0.v, false, b1.v, (short)0, acc[0][1], false, false);
        acc[1][0] = __builtin_amdgcn_wmma_f32_16x16x32_f16(
            false, a1.v, false, b0.v, (short)0, acc[1][0], false, false);
        acc[1][1] = __builtin_amdgcn_wmma_f32_16x16x32_f16(
            false, a1.v, false, b1.v, (short)0, acc[1][1], false, false);
        __syncthreads();
    }

    // ---- epilogue: C/D layout: lane n=lane%16, row m = r + (lane>=16 ? 8 : 0)
    const int mh = (lane & 16) ? 8 : 0;
    #pragma unroll
    for (int ti = 0; ti < 2; ++ti) {
        #pragma unroll
        for (int tj = 0; tj < 2; ++tj) {
            #pragma unroll
            for (int r = 0; r < 8; ++r) {
                int m = m0 + wm + ti * 16 + r + mh;
                int n = n0 + wn + tj * 16 + (lane & 15);
                float v = acc[ti][tj][r] + bias[n];
                if (MODE == 0) {
                    v = 0.5f * v * (1.0f + erff(v * 0.70710678118654752f));
                    Ch[(size_t)m * FF + n] = (_Float16)v;
                } else {
                    int bb = m >> 10;          // m = ((b*32+h)*32+w)
                    int hw = m & 1023;
                    size_t xi = ((size_t)bb * DIM + n) * (HW * HW) + hw;
                    Cf[xi] = x_orig[xi] + gamma[n] * v;
                }
            }
        }
    }
}

// ---------------------------------------------------------------------------
// Host-side launch
// ---------------------------------------------------------------------------
extern "C" void kernel_launch(void* const* d_in, const int* in_sizes, int n_in,
                              void* d_out, int out_size, void* d_ws, size_t ws_size,
                              hipStream_t stream) {
    (void)in_sizes; (void)n_in; (void)out_size; (void)ws_size;
    const float* x      = (const float*)d_in[0];
    const float* pe_w1  = (const float*)d_in[1];
    const float* pe_bng = (const float*)d_in[2];
    const float* pe_bnb = (const float*)d_in[3];
    const float* pe_bnm = (const float*)d_in[4];
    const float* pe_bnv = (const float*)d_in[5];
    const float* pe_w2  = (const float*)d_in[6];
    const float* pe_b2  = (const float*)d_in[7];
    const float* kg_w1  = (const float*)d_in[8];
    const float* kg_bng = (const float*)d_in[9];
    const float* kg_bnb = (const float*)d_in[10];
    const float* kg_bnm = (const float*)d_in[11];
    const float* kg_bnv = (const float*)d_in[12];
    const float* kg_w2  = (const float*)d_in[13];
    const float* kg_b2  = (const float*)d_in[14];
    const float* bias   = (const float*)d_in[15];
    const float* ln_w   = (const float*)d_in[16];
    const float* ln_b   = (const float*)d_in[17];
    const float* pw1_w  = (const float*)d_in[18];
    const float* pw1_b  = (const float*)d_in[19];
    const float* pw2_w  = (const float*)d_in[20];
    const float* pw2_b  = (const float*)d_in[21];
    const float* gamma  = (const float*)d_in[22];

    // Workspace layout (bytes, 256-aligned).  y_nhwc aliases kw (kw is dead
    // after rowcol_mean; stream order guarantees safety & determinism).
    char* ws = (char*)d_ws;
    float*    x1    = (float*)   (ws + 0);          // 33,554,432 B
    float*    kw    = (float*)   (ws + 33554432);   // 33,554,432 B
    float*    ynhwc = kw;                           // alias
    float*    hmid  = (float*)   (ws + 67108864);   //  2,097,152 B
    float*    Hw    = (float*)   (ws + 69206016);   //    524,288 B
    float*    Ww    = (float*)   (ws + 69730304);   //    524,288 B
    _Float16* aln   = (_Float16*)(ws + 70254592);   // 16,777,216 B
    _Float16* w1h   = (_Float16*)(ws + 87031808);   //  2,097,152 B
    _Float16* w2h   = (_Float16*)(ws + 89128960);   //  2,097,152 B
    _Float16* z1h   = (_Float16*)(ws + 91226112);   // 67,108,864 B

    // Stage 1: x1 = x + gen_pe(x)
    conv_reduce_bn_relu<<<BATCH * HW, 256, 0, stream>>>(
        x, pe_w1, pe_bng, pe_bnb, pe_bnm, pe_bnv, hmid);
    conv_expand<<<BATCH * HW * 8, 256, 0, stream>>>(
        hmid, pe_w2, pe_b2, x, x1);

    // Stage 2: kw = gen_kg(x1)
    conv_reduce_bn_relu<<<BATCH * HW, 256, 0, stream>>>(
        x1, kg_w1, kg_bng, kg_bnb, kg_bnm, kg_bnv, hmid);
    conv_expand<<<BATCH * HW * 8, 256, 0, stream>>>(
        hmid, kg_w2, kg_b2, nullptr, kw);

    // Stage 3: row/col means -> circulant kernels
    rowcol_mean<<<1024, 256, 0, stream>>>(kw, Hw, Ww);

    // Stage 4: circulant mixing + bias -> NHWC fp32 (aliases kw)
    circulant_mix<<<8388608 / 256, 256, 0, stream>>>(x1, Hw, Ww, bias, ynhwc);

    // Stage 5: LayerNorm -> f16 activations
    layernorm_f16<<<NPIX / 8, 256, 0, stream>>>(ynhwc, ln_w, ln_b, aln);

    // Stage 6: weight conversion (layouts already n-major for our GEMM)
    cvt_f32_to_f16<<<(FF * DIM) / 256, 256, 0, stream>>>(pw1_w, w1h, FF * DIM);
    cvt_f32_to_f16<<<(FF * DIM) / 256, 256, 0, stream>>>(pw2_w, w2h, FF * DIM);

    // Stage 7: MLP via WMMA.  GEMM1: 16384x2048 (K=512) + GELU -> z1h (f16)
    gemm_wmma<DIM, 0><<<dim3(FF / 128, NPIX / 64), 256, 0, stream>>>(
        aln, w1h, pw1_b, nullptr, nullptr, z1h, nullptr);
    // GEMM2: 16384x512 (K=2048), fused +pw2_b, *gamma, +x residual -> d_out
    gemm_wmma<FF, 1><<<dim3(DIM / 128, NPIX / 64), 256, 0, stream>>>(
        z1h, w2h, pw2_b, x, gamma, nullptr, (float*)d_out);
}
